// MultiHeadAttention_2783138808516
// MI455X (gfx1250) — matrix operand order repaired and checked by reading, not compiled, not run on hardware
//
#include <hip/hip_runtime.h>

#define DMODEL 1024
#define HEADS  16
#define HDIM   64
#define BATCH  2
#define SEQ    2048

typedef __attribute__((ext_vector_type(16))) __bf16 v16bf;
typedef __attribute__((ext_vector_type(8)))  float  v8f;

union FragBF { v16bf v; uint4 u[2]; __bf16 e[16]; };
union BF8U   { uint4 u; __bf16 e[8]; };

__device__ inline v8f v8zero() {
  v8f z;
#pragma unroll
  for (int i = 0; i < 8; ++i) z[i] = 0.0f;
  return z;
}

__device__ inline v8f wmma_bf16(v16bf a, v16bf b, v8f c) {
  // D = A(16x32 bf16) * B(32x16 bf16) + C(16x16 f32)
  return __builtin_amdgcn_wmma_f32_16x16x32_bf16(
      /*neg_a=*/false, a, /*neg_b=*/false, b,
      /*c_mod=*/(short)0, c, /*reuse_a=*/false, /*reuse_b=*/false);
}

// Async global -> LDS copy of 16 bytes per lane (ASYNCcnt-tracked, GVS mode).
__device__ inline void async_copy_b128(unsigned ldsAddr, unsigned byteOff,
                                       const void* base) {
  asm volatile("global_load_async_to_lds_b128 %0, %1, %2"
               :
               : "v"(ldsAddr), "v"(byteOff),
                 "s"((unsigned long long)(uintptr_t)base)
               : "memory");
}
__device__ inline void wait_async() {
  asm volatile("s_wait_asynccnt 0" ::: "memory");
}

__device__ inline float redmax16(float v) {
#pragma unroll
  for (int m = 8; m >= 1; m >>= 1) v = fmaxf(v, __shfl_xor(v, m, 32));
  return v;
}
__device__ inline float redsum16(float v) {
#pragma unroll
  for (int m = 8; m >= 1; m >>= 1) v += __shfl_xor(v, m, 32);
  return v;
}

// ---------------------------------------------------------------------------
// f32 -> bf16 conversion (4 elems / thread, 8-byte stores)
// ---------------------------------------------------------------------------
__global__ __launch_bounds__(256)
void cvt_f32_bf16(const float* __restrict__ in, __bf16* __restrict__ out, int n) {
  int i = (blockIdx.x * 256 + threadIdx.x) * 4;
  if (i + 3 < n) {
    float4 f = *(const float4*)(in + i);
    union { __bf16 e[4]; unsigned long long u; } p;
    p.e[0] = (__bf16)f.x; p.e[1] = (__bf16)f.y;
    p.e[2] = (__bf16)f.z; p.e[3] = (__bf16)f.w;
    *(unsigned long long*)(out + i) = p.u;
  }
}

// ---------------------------------------------------------------------------
// NT GEMM: C[M,N] = A[M,K] * W[N,K]^T   (bf16 in, f32 accum)
// MODE 0: bf16 out, scattered to [B,H,S,HD] head layout (QKV projections)
// MODE 1: f32 out + bias, row-major [M,N]  (output projection)
// Block tile 128x128, BK=32, 8 waves each computing 64x32.
// Tiles staged with gfx1250 async global->LDS copies (no VGPR staging).
// ---------------------------------------------------------------------------
template <int MODE>
__global__ __launch_bounds__(256)
void gemm_nt_bf16(const __bf16* __restrict__ A, const __bf16* __restrict__ W,
                  const float* __restrict__ bias, void* __restrict__ outp) {
  constexpr int K  = DMODEL;
  constexpr int BK = 32;
  __shared__ __bf16 lA[128][BK + 8];
  __shared__ __bf16 lB[128][BK + 8];

  const int t    = threadIdx.x;
  const int wave = t >> 5, lane = t & 31;
  const int wm   = wave & 1;    // 2 waves along M
  const int wn   = wave >> 1;   // 4 waves along N
  const int m0   = blockIdx.y * 128;
  const int n0   = blockIdx.x * 128;
  const int mIdx = lane & 15, hlf = lane >> 4;

  v8f acc[4][2];
#pragma unroll
  for (int fm = 0; fm < 4; ++fm)
#pragma unroll
    for (int fn = 0; fn < 2; ++fn) acc[fm][fn] = v8zero();

  for (int kt = 0; kt < K / BK; ++kt) {
    const int kbase = kt * BK;
#pragma unroll
    for (int c = 0; c < 2; ++c) {
      int idx = t + c * 256;              // 512 chunks of 8 bf16
      int row = idx >> 2, seg = idx & 3;
      unsigned la   = (unsigned)(uintptr_t)&lA[row][seg * 8];
      unsigned lb   = (unsigned)(uintptr_t)&lB[row][seg * 8];
      unsigned offA = (unsigned)(((m0 + row) * K + kbase + seg * 8) * 2);
      unsigned offB = (unsigned)(((n0 + row) * K + kbase + seg * 8) * 2);
      async_copy_b128(la, offA, A);
      async_copy_b128(lb, offB, W);
    }
    wait_async();
    __syncthreads();

    if (kt + 1 < K / BK) {  // gfx1250 global_prefetch_b8 for the next tile
      __builtin_prefetch(A + (size_t)(m0 + (t >> 2)) * K + kbase + BK + (t & 3) * 8, 0, 0);
      __builtin_prefetch(W + (size_t)(n0 + (t >> 2)) * K + kbase + BK + (t & 3) * 8, 0, 0);
    }

    FragBF fa[4], fb[2];
#pragma unroll
    for (int fm = 0; fm < 4; ++fm) {
      const __bf16* p = &lA[wm * 64 + fm * 16 + mIdx][hlf * 8];
      fa[fm].u[0] = *(const uint4*)p;          // K = hlf*8 .. +7
      fa[fm].u[1] = *(const uint4*)(p + 16);   // K = hlf*8+16 .. +7
    }
#pragma unroll
    for (int fn = 0; fn < 2; ++fn) {
      const __bf16* p = &lB[wn * 32 + fn * 16 + mIdx][hlf * 8];
      fb[fn].u[0] = *(const uint4*)p;          // column mIdx, K = hlf*8 .. +7
      fb[fn].u[1] = *(const uint4*)(p + 16);   // K = hlf*8+16 .. +7
    }
#pragma unroll
    for (int fm = 0; fm < 4; ++fm)
#pragma unroll
      for (int fn = 0; fn < 2; ++fn)
        acc[fm][fn] = wmma_bf16(fa[fm].v, fb[fn].v, acc[fm][fn]);
    __syncthreads();
  }

#pragma unroll
  for (int fm = 0; fm < 4; ++fm)
#pragma unroll
    for (int fn = 0; fn < 2; ++fn)
#pragma unroll
      for (int r = 0; r < 8; ++r) {
        int gm = m0 + wm * 64 + fm * 16 + r + 8 * hlf;
        int gn = n0 + wn * 32 + fn * 16 + mIdx;
        if constexpr (MODE == 0) {
          __bf16* o = (__bf16*)outp;
          int bb = gm >> 11, s = gm & (SEQ - 1);
          int hh = gn >> 6,  d = gn & (HDIM - 1);
          o[(((size_t)bb * HEADS + hh) * SEQ + s) * HDIM + d] = (__bf16)acc[fm][fn][r];
        } else {
          float* o = (float*)outp;
          o[(size_t)gm * DMODEL + gn] = acc[fm][fn][r] + bias[gn];
        }
      }
}

// ---------------------------------------------------------------------------
// Flash attention: one workgroup = (b, h, 128 q rows); 8 waves x 16 rows.
// K-loop over 64-key tiles with online softmax; mask applied pre-scale.
// K tile staged via async global->LDS copy; V staged+transposed via VGPRs.
// ---------------------------------------------------------------------------
__global__ __launch_bounds__(256)
void attn_kernel(const __bf16* __restrict__ Qh, const __bf16* __restrict__ Kh,
                 const __bf16* __restrict__ Vh, const int* __restrict__ mask,
                 __bf16* __restrict__ attn) {
  __shared__ __bf16 lK[64][72];      // [key][hd]
  __shared__ __bf16 lVt[64][72];     // [hd][key]  (transposed on fill)
  __shared__ __bf16 lP[8][16][72];   // per-wave P relayout scratch

  const int t    = threadIdx.x;
  const int w    = t >> 5, lane = t & 31;
  const int nIdx = lane & 15, hlf = lane >> 4;
  const int q0   = blockIdx.x * 128;
  const int h    = blockIdx.y, b = blockIdx.z;

  const size_t headBase = ((size_t)(b * HEADS + h)) * SEQ * HDIM;

  // Q A-fragments, loaded once from global (contiguous 16B per lane)
  FragBF fq[2];
  {
    const __bf16* qrow = Qh + headBase + (size_t)(q0 + w * 16 + nIdx) * HDIM;
#pragma unroll
    for (int kk = 0; kk < 2; ++kk) {
      fq[kk].u[0] = *(const uint4*)(qrow + kk * 32 + hlf * 8);
      fq[kk].u[1] = *(const uint4*)(qrow + kk * 32 + hlf * 8 + 16);
    }
  }

  v8f accO[4];
#pragma unroll
  for (int fn = 0; fn < 4; ++fn) accO[fn] = v8zero();
  float m_i[8], l_i[8];
#pragma unroll
  for (int r = 0; r < 8; ++r) { m_i[r] = -3.0e38f; l_i[r] = 0.0f; }

  for (int kb = 0; kb < SEQ / 64; ++kb) {
    const size_t kvbase = headBase + (size_t)kb * 64 * HDIM;  // elements
#pragma unroll
    for (int c = 0; c < 2; ++c) {
      int idx = t + c * 256;             // 512 chunks of 8 bf16
      int row = idx >> 3, seg = idx & 7;
      // K: async global->LDS (no VGPR staging)
      unsigned lk   = (unsigned)(uintptr_t)&lK[row][seg * 8];
      unsigned offK = (unsigned)((kvbase + (size_t)row * HDIM + seg * 8) * 2);
      async_copy_b128(lk, offK, Kh);
      // V: VGPR-staged, scattered transposed into lVt
      BF8U vv;
      vv.u = *(const uint4*)(Vh + kvbase + (size_t)row * HDIM + seg * 8);
#pragma unroll
      for (int e = 0; e < 8; ++e) lVt[seg * 8 + e][row] = vv.e[e];
    }
    wait_async();
    __syncthreads();

    // S = Q * K^T  (16 q rows x 64 keys per wave)
    v8f sc[4];
#pragma unroll
    for (int fn = 0; fn < 4; ++fn) sc[fn] = v8zero();
#pragma unroll
    for (int fn = 0; fn < 4; ++fn)
#pragma unroll
      for (int kk = 0; kk < 2; ++kk) {
        FragBF kbf;
        const __bf16* p = &lK[fn * 16 + nIdx][kk * 32 + hlf * 8];
        kbf.u[0] = *(const uint4*)p;
        kbf.u[1] = *(const uint4*)(p + 16);
        sc[fn] = wmma_bf16(fq[kk].v, kbf.v, sc[fn]);
      }

    // mask (pre-scale, like the reference), scale, online softmax
    float rm[8];
#pragma unroll
    for (int r = 0; r < 8; ++r) rm[r] = -3.0e38f;
#pragma unroll
    for (int fn = 0; fn < 4; ++fn)
#pragma unroll
      for (int r = 0; r < 8; ++r) {
        int q    = q0 + w * 16 + r + 8 * hlf;
        int kcol = kb * 64 + fn * 16 + nIdx;
        int mv   = mask[(size_t)b * SEQ * SEQ + (size_t)q * SEQ + kcol];
        float s  = (mv == 0 ? -1.0e20f : sc[fn][r]) * 0.125f;  // 1/sqrt(64)
        sc[fn][r] = s;
        rm[r] = fmaxf(rm[r], s);
      }
#pragma unroll
    for (int r = 0; r < 8; ++r) rm[r] = redmax16(rm[r]);
    float alpha[8];
#pragma unroll
    for (int r = 0; r < 8; ++r) {
      float nm = fmaxf(m_i[r], rm[r]);
      alpha[r] = __expf(m_i[r] - nm);
      m_i[r]   = nm;
    }
    float rs[8];
#pragma unroll
    for (int r = 0; r < 8; ++r) rs[r] = 0.0f;
#pragma unroll
    for (int fn = 0; fn < 4; ++fn)
#pragma unroll
      for (int r = 0; r < 8; ++r) {
        float e = __expf(sc[fn][r] - m_i[r]);
        sc[fn][r] = e;
        rs[r] += e;
      }
#pragma unroll
    for (int r = 0; r < 8; ++r) {
      rs[r]  = redsum16(rs[r]);
      l_i[r] = l_i[r] * alpha[r] + rs[r];
    }
#pragma unroll
    for (int fn = 0; fn < 4; ++fn)
#pragma unroll
      for (int r = 0; r < 8; ++r) accO[fn][r] *= alpha[r];

    // C-layout -> A-layout for P via per-wave LDS scratch
#pragma unroll
    for (int fn = 0; fn < 4; ++fn)
#pragma unroll
      for (int r = 0; r < 8; ++r)
        lP[w][r + 8 * hlf][fn * 16 + nIdx] = (__bf16)sc[fn][r];
    asm volatile("s_wait_dscnt 0" ::: "memory");
    FragBF fp[2];
#pragma unroll
    for (int kk = 0; kk < 2; ++kk) {
      const __bf16* p = &lP[w][nIdx][kk * 32 + hlf * 8];
      fp[kk].u[0] = *(const uint4*)p;
      fp[kk].u[1] = *(const uint4*)(p + 16);
    }

    // O += P * V   (B-frag = contiguous row of transposed V)
#pragma unroll
    for (int fn = 0; fn < 4; ++fn)
#pragma unroll
      for (int kk = 0; kk < 2; ++kk) {
        FragBF vb;
        const __bf16* p = &lVt[fn * 16 + nIdx][kk * 32 + hlf * 8];
        vb.u[0] = *(const uint4*)p;
        vb.u[1] = *(const uint4*)(p + 16);
        accO[fn] = wmma_bf16(fp[kk].v, vb.v, accO[fn]);
      }
    __syncthreads();
  }

  // normalize and scatter back to [B, S, H*HD] bf16 for the output projection
  float inv[8];
#pragma unroll
  for (int r = 0; r < 8; ++r) inv[r] = 1.0f / l_i[r];
#pragma unroll
  for (int fn = 0; fn < 4; ++fn)
#pragma unroll
    for (int r = 0; r < 8; ++r) {
      int q = q0 + w * 16 + r + 8 * hlf;
      attn[((size_t)b * SEQ + q) * DMODEL + h * HDIM + fn * 16 + nIdx] =
          (__bf16)(accO[fn][r] * inv[r]);
    }
}

// ---------------------------------------------------------------------------
// Launch
// ---------------------------------------------------------------------------
extern "C" void kernel_launch(void* const* d_in, const int* in_sizes, int n_in,
                              void* d_out, int out_size, void* d_ws, size_t ws_size,
                              hipStream_t stream) {
  (void)in_sizes; (void)n_in; (void)out_size; (void)ws_size;

  const float* key   = (const float*)d_in[0];
  const float* query = (const float*)d_in[1];
  const float* value = (const float*)d_in[2];
  const int*   mask  = (const int*)d_in[3];
  const float* Wq    = (const float*)d_in[4];
  const float* Wk    = (const float*)d_in[5];
  const float* Wv    = (const float*)d_in[6];
  const float* Wo    = (const float*)d_in[7];
  const float* bo    = (const float*)d_in[8];
  float*       out   = (float*)d_out;

  constexpr size_t NX = (size_t)BATCH * SEQ * DMODEL;  // 4194304
  constexpr size_t NW = (size_t)DMODEL * DMODEL;       // 1048576

  __bf16* ws  = (__bf16*)d_ws;
  __bf16* xq  = ws;
  __bf16* xk  = xq + NX;
  __bf16* xv  = xk + NX;
  __bf16* wq  = xv + NX;
  __bf16* wk  = wq + NW;
  __bf16* wv  = wk + NW;
  __bf16* wo  = wv + NW;
  __bf16* Qh  = wo + NW;
  __bf16* Kh  = Qh + NX;
  __bf16* Vh  = Kh + NX;
  __bf16* atn = Vh + NX;

  const int gX = (int)(NX / 1024);  // 256 thr * 4 elem
  const int gW = (int)(NW / 1024);
  cvt_f32_bf16<<<gX, 256, 0, stream>>>(query, xq, (int)NX);
  cvt_f32_bf16<<<gX, 256, 0, stream>>>(key,   xk, (int)NX);
  cvt_f32_bf16<<<gX, 256, 0, stream>>>(value, xv, (int)NX);
  cvt_f32_bf16<<<gW, 256, 0, stream>>>(Wq, wq, (int)NW);
  cvt_f32_bf16<<<gW, 256, 0, stream>>>(Wk, wk, (int)NW);
  cvt_f32_bf16<<<gW, 256, 0, stream>>>(Wv, wv, (int)NW);
  cvt_f32_bf16<<<gW, 256, 0, stream>>>(Wo, wo, (int)NW);

  dim3 gg(DMODEL / 128, (BATCH * SEQ) / 128);  // (8, 32)
  gemm_nt_bf16<0><<<gg, 256, 0, stream>>>(xq, wq, nullptr, Qh);
  gemm_nt_bf16<0><<<gg, 256, 0, stream>>>(xk, wk, nullptr, Kh);
  gemm_nt_bf16<0><<<gg, 256, 0, stream>>>(xv, wv, nullptr, Vh);

  dim3 ga(SEQ / 128, HEADS, BATCH);            // (16, 16, 2)
  attn_kernel<<<ga, 256, 0, stream>>>(Qh, Kh, Vh, mask, atn);

  gemm_nt_bf16<1><<<gg, 256, 0, stream>>>(atn, wo, bo, out);
}
